// BigBirdModel_69131793596542
// MI455X (gfx1250) — matrix operand order repaired and checked by reading, not compiled, not run on hardware
//
#include <hip/hip_runtime.h>
#include <math.h>

// ---------------------------------------------------------------------------
// Types for CDNA5 WMMA (wave32)
// ---------------------------------------------------------------------------
typedef __attribute__((ext_vector_type(16))) __bf16 v16bf;
typedef __attribute__((ext_vector_type(8)))  float  v8f;
typedef __attribute__((ext_vector_type(4)))  unsigned int u32x4;

#define NEG_MASK (-10000.0f)

__device__ __forceinline__ unsigned short f2bf(float f) {
    unsigned int u = __float_as_uint(f);
    unsigned int r = u + 0x7FFFu + ((u >> 16) & 1u);   // round-to-nearest-even
    return (unsigned short)(r >> 16);
}

// Assemble a 16-element bf16 fragment from two 16-byte LDS chunks.
__device__ __forceinline__ v16bf frag_from(const unsigned short* p0,
                                           const unsigned short* p1) {
    union { v16bf v; u32x4 q[2]; } u;
    u.q[0] = *(const u32x4*)p0;
    u.q[1] = *(const u32x4*)p1;
    return u.v;
}

// ---------------------------------------------------------------------------
// CDNA5 async global->LDS copy (16B per lane), tracked by ASYNCcnt.
// LDS address = low 32 bits of the generic shared-pointer (aperture rule).
// ---------------------------------------------------------------------------
__device__ __forceinline__ void async_b128(const void* gptr, void* lptr) {
    unsigned lo = (unsigned)(size_t)lptr;
    asm volatile("global_load_async_to_lds_b128 %0, %1, off"
                 :: "v"(lo), "v"((unsigned long long)(size_t)gptr)
                 : "memory");
}
__device__ __forceinline__ void wait_async() {
    asm volatile("s_wait_asynccnt 0x0" ::: "memory");
}

// ---------------------------------------------------------------------------
// fp32 -> bf16 bit conversion (weights, once per launch)
// ---------------------------------------------------------------------------
__global__ __launch_bounds__(256) void k_convert(const float* __restrict__ s,
                                                 unsigned short* __restrict__ d,
                                                 int n) {
    int i = blockIdx.x * 256 + threadIdx.x;
    if (i < n) d[i] = f2bf(s[i]);
}

// ---------------------------------------------------------------------------
// Patch gather: pixel (2,3,512,512) -> Pm bf16 (2*1024, 768)
// ---------------------------------------------------------------------------
__global__ __launch_bounds__(256) void k_patches(const float* __restrict__ pix,
                                                 unsigned short* __restrict__ Pm) {
    int i = blockIdx.x * 256 + threadIdx.x;
    if (i >= 2 * 1024 * 768) return;
    int d   = i % 768;
    int rem = i / 768;
    int p   = rem % 1024;
    int b   = rem / 1024;
    int c  = d / 256;
    int py = (d % 256) / 16;
    int px = d % 16;
    int gy = p >> 5, gx = p & 31;
    size_t src = (size_t)b * (3 * 512 * 512) + (size_t)c * (512 * 512) +
                 (size_t)(gy * 16 + py) * 512 + (gx * 16 + px);
    Pm[i] = f2bf(pix[src]);
}

// ---------------------------------------------------------------------------
// Assemble X (2,1088,768) fp32
// ---------------------------------------------------------------------------
__global__ __launch_bounds__(256) void k_assemble(const float* __restrict__ Xp,
                                                  const float* __restrict__ cls,
                                                  const float* __restrict__ pos,
                                                  float* __restrict__ X) {
    int i = blockIdx.x * 256 + threadIdx.x;
    if (i >= 2 * 1088 * 768) return;
    int d   = i % 768;
    int rem = i / 768;
    int tok = rem % 1088;
    int b   = rem / 1088;
    float v;
    if (tok == 0)        v = cls[d] + pos[d];
    else if (tok < 1025) v = Xp[((size_t)(b * 1024 + tok - 1)) * 768 + d] +
                             pos[(size_t)tok * 768 + d];
    else                 v = 0.0f;
    X[i] = v;
}

// ---------------------------------------------------------------------------
// LayerNorm: X fp32 (2176,768) -> bf16 out, padded rows (tok>=1025) zeroed
// ---------------------------------------------------------------------------
__global__ __launch_bounds__(256) void k_ln_bf16(const float* __restrict__ X,
                                                 unsigned short* __restrict__ out,
                                                 const float* __restrict__ g,
                                                 const float* __restrict__ be) {
    int row = blockIdx.x;               // 0..2175
    int tok = row % 1088;
    int t = threadIdx.x;
    unsigned short* o = out + (size_t)row * 768;
    if (tok >= 1025) { o[t] = 0; o[t + 256] = 0; o[t + 512] = 0; return; }
    const float* x = X + (size_t)row * 768;
    float v0 = x[t], v1 = x[t + 256], v2 = x[t + 512];
    __shared__ float red[256];
    red[t] = v0 + v1 + v2;
    __syncthreads();
    for (int w = 128; w > 0; w >>= 1) { if (t < w) red[t] += red[t + w]; __syncthreads(); }
    float mean = red[0] * (1.0f / 768.0f);
    __syncthreads();
    float d0 = v0 - mean, d1 = v1 - mean, d2 = v2 - mean;
    red[t] = d0 * d0 + d1 * d1 + d2 * d2;
    __syncthreads();
    for (int w = 128; w > 0; w >>= 1) { if (t < w) red[t] += red[t + w]; __syncthreads(); }
    float rstd = rsqrtf(red[0] * (1.0f / 768.0f) + 1e-5f);
    o[t]       = f2bf(d0 * rstd * g[t]       + be[t]);
    o[t + 256] = f2bf(d1 * rstd * g[t + 256] + be[t + 256]);
    o[t + 512] = f2bf(d2 * rstd * g[t + 512] + be[t + 512]);
}

// Final LayerNorm: padded X -> compact fp32 output (2*1025,768)
__global__ __launch_bounds__(256) void k_ln_out(const float* __restrict__ X,
                                                float* __restrict__ out,
                                                const float* __restrict__ g,
                                                const float* __restrict__ be) {
    int row = blockIdx.x;               // 0..2049
    int b = row / 1025, tok = row % 1025;
    int t = threadIdx.x;
    const float* x = X + ((size_t)(b * 1088 + tok)) * 768;
    float* o = out + (size_t)row * 768;
    float v0 = x[t], v1 = x[t + 256], v2 = x[t + 512];
    __shared__ float red[256];
    red[t] = v0 + v1 + v2;
    __syncthreads();
    for (int w = 128; w > 0; w >>= 1) { if (t < w) red[t] += red[t + w]; __syncthreads(); }
    float mean = red[0] * (1.0f / 768.0f);
    __syncthreads();
    float d0 = v0 - mean, d1 = v1 - mean, d2 = v2 - mean;
    red[t] = d0 * d0 + d1 * d1 + d2 * d2;
    __syncthreads();
    for (int w = 128; w > 0; w >>= 1) { if (t < w) red[t] += red[t + w]; __syncthreads(); }
    float rstd = rsqrtf(red[0] * (1.0f / 768.0f) + 1e-5f);
    o[t]       = d0 * rstd * g[t]       + be[t];
    o[t + 256] = d1 * rstd * g[t + 256] + be[t + 256];
    o[t + 512] = d2 * rstd * g[t + 512] + be[t + 512];
}

// ---------------------------------------------------------------------------
// Tiled WMMA GEMM:  C(MxN) = A(MxK) @ W(KxN) + bias  [+gelu] [+residual]
// Block tile 128x128, BK=32, 8 waves, each wave 64x32 (4x2 wmma frags).
// A tile staged via async global->LDS (bf16 path) or register-convert (fp32).
// B tile staged transposed [n][k] via registers (needed for B-operand layout).
// M % 128 == 0, N % 128 == 0, K % 32 == 0 (true for all uses).
// ---------------------------------------------------------------------------
template <bool A_F32, bool OUT_BF16, bool DO_GELU, bool RESID>
__global__ __launch_bounds__(256) void k_gemm(const void* __restrict__ Ap,
                                              const unsigned short* __restrict__ Bw,
                                              const float* __restrict__ bias,
                                              float* __restrict__ Cf,
                                              unsigned short* __restrict__ Cb,
                                              int M, int N, int K) {
    __shared__ __align__(16) unsigned short sA[128 * 32];
    __shared__ __align__(16) unsigned short sB[128 * 32];  // transposed: [n][k]
    const int tid  = threadIdx.x;
    const int lane = tid & 31;
    const int wid  = tid >> 5;
    const int half = lane >> 4, ln16 = lane & 15;
    const int bm = blockIdx.y, bn = blockIdx.x;
    const int row0 = bm * 128, col0 = bn * 128;
    const int mb  = (wid >> 2) * 64;    // 2 wave-rows of 64
    const int nb2 = (wid & 3) * 32;     // 4 wave-cols of 32

    v8f zf = {};
    v8f acc[4][2];
    #pragma unroll
    for (int tm = 0; tm < 4; tm++) { acc[tm][0] = zf; acc[tm][1] = zf; }

    const float*          Af = (const float*)Ap;
    const unsigned short* Ab = (const unsigned short*)Ap;

    for (int k0 = 0; k0 < K; k0 += 32) {
        // ---- A tile 128x32 (row-major in LDS) ----
        if constexpr (!A_F32) {
            // async DMA: 512 chunks of 16B, 2 per thread
            #pragma unroll
            for (int i = 0; i < 2; i++) {
                int c  = i * 256 + tid;
                int r  = c >> 2;             // 4 chunks per 32-elem row
                int o8 = (c & 3) * 8;
                async_b128(Ab + (size_t)(row0 + r) * K + k0 + o8,
                           sA + r * 32 + o8);
            }
        } else {
            #pragma unroll
            for (int i = 0; i < 16; i++) {
                int l = i * 256 + tid;
                int r = l >> 5, c = l & 31;
                sA[r * 32 + c] = f2bf(Af[(size_t)(row0 + r) * K + k0 + c]);
            }
        }
        // ---- B tile 32x128, stored transposed [n][k] ----
        #pragma unroll
        for (int i = 0; i < 16; i++) {
            int l = i * 256 + tid;
            int kk = l >> 7, n = l & 127;
            sB[n * 32 + kk] = Bw[(size_t)(k0 + kk) * N + col0 + n];
        }
        // prefetch next B tile into caches (global_prefetch_b8)
        if (k0 + 32 < K)
            __builtin_prefetch(Bw + (size_t)(k0 + 32) * N + col0 + tid * 16, 0, 0);

        if constexpr (!A_F32) wait_async();
        __syncthreads();

        v16bf afr[4], bfr[2];
        #pragma unroll
        for (int tm = 0; tm < 4; tm++) {
            // A layout: lanes 0-15 rows M, elems K{0..7,16..23}; lanes16-31 K{8..15,24..31}
            const unsigned short* pa = sA + (size_t)(mb + tm * 16 + ln16) * 32 + (half ? 8 : 0);
            afr[tm] = frag_from(pa, pa + 16);
        }
        #pragma unroll
        for (int tn = 0; tn < 2; tn++) {
            // B layout: lane n = column, elems K{0..7,16..23} (lanes 0-15) / K{8..15,24..31} (16-31)
            const unsigned short* pb = sB + (size_t)(nb2 + tn * 16 + ln16) * 32 + (half ? 8 : 0);
            bfr[tn] = frag_from(pb, pb + 16);
        }
        #pragma unroll
        for (int tm = 0; tm < 4; tm++)
            #pragma unroll
            for (int tn = 0; tn < 2; tn++)
                acc[tm][tn] = __builtin_amdgcn_wmma_f32_16x16x32_bf16(
                    false, afr[tm], false, bfr[tn], (short)0, acc[tm][tn], false, false);
        __syncthreads();
    }

    // Epilogue: bias [+gelu] [+residual] ; C layout: VGPR r -> rows r / 8+r
    #pragma unroll
    for (int tm = 0; tm < 4; tm++)
        #pragma unroll
        for (int tn = 0; tn < 2; tn++)
            #pragma unroll
            for (int r = 0; r < 8; r++) {
                int m   = half ? (8 + r) : r;
                int row = row0 + mb + tm * 16 + m;
                int col = col0 + nb2 + tn * 16 + ln16;
                float v = acc[tm][tn][r] + bias[col];
                if constexpr (DO_GELU)
                    v = 0.5f * v * (1.0f + erff(v * 0.70710678118654752f));
                size_t oi = (size_t)row * N + col;
                if constexpr (RESID)          Cf[oi] += v;
                else if constexpr (OUT_BF16)  Cb[oi] = f2bf(v);
                else                          Cf[oi] = v;
            }
}

// ---------------------------------------------------------------------------
// BigBird sparse attention, flash-style streaming softmax.
// Grid (17, 12, 2), block 128 (4 waves); each wave owns 16 query rows.
// Q/K tiles staged via async global->LDS DMA; V transposed via registers.
// ---------------------------------------------------------------------------
__global__ __launch_bounds__(128) void k_attn(const unsigned short* __restrict__ Q,
                                              const unsigned short* __restrict__ K,
                                              const unsigned short* __restrict__ V,
                                              const int* __restrict__ ra,
                                              float* __restrict__ ctx) {
    const int qb = blockIdx.x, h = blockIdx.y, b = blockIdx.z;
    const int tid  = threadIdx.x;
    const int lane = tid & 31, wave = tid >> 5;
    const int half = lane >> 4, ln16 = lane & 15;

    __shared__ __align__(16) unsigned short sQ[64 * 64];
    __shared__ __align__(16) unsigned short sK[64 * 64];   // [key][hd]
    __shared__ __align__(16) unsigned short sVt[64 * 64];  // [hd][key]
    __shared__ __align__(16) unsigned short sP[64 * 64];   // probabilities bf16
    __shared__ int slist[17];
    __shared__ int snb;

    if (tid == 0) {
        int c = 0;
        if (qb == 0 || qb == 16) {
            for (int i = 0; i < 17; i++) slist[c++] = i;    // full attention
        } else {
            slist[c++] = 0;                                  // global first block
            if (qb == 1)       { slist[c++] = 1;  slist[c++] = 2; }
            else if (qb == 15) { slist[c++] = 13; slist[c++] = 14; slist[c++] = 15; }
            else               { slist[c++] = qb - 1; slist[c++] = qb; slist[c++] = qb + 1; }
            slist[c++] = 16;                                 // global last block
            int m = qb - 1;
            for (int j = 0; j < 3; j++) slist[c++] = ra[(h * 15 + m) * 3 + j];
        }
        snb = c;
    }

    const size_t basebh = (size_t)b * 1088 * 768 + (size_t)h * 64;
    // async DMA Q block: 512 x 16B chunks, 4 per thread
    #pragma unroll
    for (int i = 0; i < 4; i++) {
        int c  = i * 128 + tid;
        int r  = c >> 3;                 // 8 chunks per 64-elem row
        int o8 = (c & 7) * 8;
        async_b128(Q + basebh + (size_t)(qb * 64 + r) * 768 + o8,
                   sQ + r * 64 + o8);
    }
    wait_async();
    __syncthreads();
    const int nb = snb;

    float rm[8], rl[8];
    v8f zf = {};
    v8f Oacc[4];
    #pragma unroll
    for (int r = 0; r < 8; r++) { rm[r] = -1e30f; rl[r] = 0.0f; }
    #pragma unroll
    for (int t = 0; t < 4; t++) Oacc[t] = zf;

    for (int ib = 0; ib < nb; ib++) {
        const int kblk = slist[ib];
        // K tile: layout-preserving copy -> async DMA
        #pragma unroll
        for (int i = 0; i < 4; i++) {
            int c  = i * 128 + tid;
            int r  = c >> 3;
            int o8 = (c & 7) * 8;
            async_b128(K + basebh + (size_t)(kblk * 64 + r) * 768 + o8,
                       sK + r * 64 + o8);
        }
        // V tile: transpose via registers
        for (int i = tid; i < 64 * 64; i += 128) {
            int r = i >> 6, cdim = i & 63;
            sVt[cdim * 64 + r] = V[basebh + (size_t)(kblk * 64 + r) * 768 + cdim];
        }
        wait_async();
        __syncthreads();

        // S = Q @ K^T  (16x64 per wave, K-dim = HD = 64)
        v8f sacc[4];
        #pragma unroll
        for (int t = 0; t < 4; t++) sacc[t] = zf;
        #pragma unroll
        for (int ks = 0; ks < 64; ks += 32) {
            const unsigned short* pa = sQ + (size_t)(wave * 16 + ln16) * 64 + ks + (half ? 8 : 0);
            v16bf aq = frag_from(pa, pa + 16);
            #pragma unroll
            for (int t = 0; t < 4; t++) {
                const unsigned short* pb = sK + (size_t)(t * 16 + ln16) * 64 + ks + (half ? 8 : 0);
                v16bf bk = frag_from(pb, pb + 16);
                sacc[t] = __builtin_amdgcn_wmma_f32_16x16x32_bf16(
                    false, aq, false, bk, (short)0, sacc[t], false, false);
            }
        }

        // scale + padding mask (only last block can contain tokens >= 1025)
        float neg[4];
        #pragma unroll
        for (int t = 0; t < 4; t++) {
            int keytok = kblk * 64 + t * 16 + ln16;
            neg[t] = (keytok >= 1025) ? NEG_MASK : 0.0f;
        }

        // online softmax update per query row
        float nm[8];
        #pragma unroll
        for (int r = 0; r < 8; r++) {
            float mx = -1e30f;
            #pragma unroll
            for (int t = 0; t < 4; t++) {
                float s = sacc[t][r] * 0.125f + neg[t];
                sacc[t][r] = s;
                mx = fmaxf(mx, s);
            }
            for (int m2 = 1; m2 < 16; m2 <<= 1) mx = fmaxf(mx, __shfl_xor(mx, m2, 32));
            nm[r] = fmaxf(rm[r], mx);
        }
        #pragma unroll
        for (int r = 0; r < 8; r++) {
            float rs = 0.0f;
            #pragma unroll
            for (int t = 0; t < 4; t++) {
                float p = __expf(sacc[t][r] - nm[r]);
                sacc[t][r] = p;
                rs += p;
            }
            for (int m2 = 1; m2 < 16; m2 <<= 1) rs += __shfl_xor(rs, m2, 32);
            float alpha = __expf(rm[r] - nm[r]);
            rl[r] = rl[r] * alpha + rs;
            rm[r] = nm[r];
            #pragma unroll
            for (int t = 0; t < 4; t++) Oacc[t][r] *= alpha;
        }

        // P -> LDS (bf16) so it can be reloaded in A-fragment layout
        #pragma unroll
        for (int t = 0; t < 4; t++)
            #pragma unroll
            for (int r = 0; r < 8; r++) {
                int m = half ? (8 + r) : r;
                sP[(wave * 16 + m) * 64 + t * 16 + ln16] = f2bf(sacc[t][r]);
            }
        __syncthreads();

        // O += P @ V   (K-dim = 64 keys, N = HD via transposed V)
        #pragma unroll
        for (int ks = 0; ks < 64; ks += 32) {
            const unsigned short* pa = sP + (size_t)(wave * 16 + ln16) * 64 + ks + (half ? 8 : 0);
            v16bf ap = frag_from(pa, pa + 16);
            #pragma unroll
            for (int t = 0; t < 4; t++) {
                const unsigned short* pb = sVt + (size_t)(t * 16 + ln16) * 64 + ks + (half ? 8 : 0);
                v16bf bv = frag_from(pb, pb + 16);
                Oacc[t] = __builtin_amdgcn_wmma_f32_16x16x32_bf16(
                    false, ap, false, bv, (short)0, Oacc[t], false, false);
            }
        }
        __syncthreads();
    }

    // finalize: divide by denom, apply from_mask (padded queries -> 0)
    #pragma unroll
    for (int t = 0; t < 4; t++)
        #pragma unroll
        for (int r = 0; r < 8; r++) {
            int m    = half ? (8 + r) : r;
            int qtok = qb * 64 + wave * 16 + m;
            float o  = (qtok < 1025) ? (Oacc[t][r] / rl[r]) : 0.0f;
            ctx[((size_t)(b * 1088 + qtok)) * 768 + h * 64 + t * 16 + ln16] = o;
        }
}

// ---------------------------------------------------------------------------
// Orchestration
// ---------------------------------------------------------------------------
extern "C" void kernel_launch(void* const* d_in, const int* in_sizes, int n_in,
                              void* d_out, int out_size, void* d_ws, size_t ws_size,
                              hipStream_t stream) {
    (void)in_sizes; (void)n_in; (void)out_size; (void)ws_size;

    const float* pixel   = (const float*)d_in[0];
    const float* patch_w = (const float*)d_in[1];
    const float* patch_b = (const float*)d_in[2];
    const float* cls     = (const float*)d_in[3];
    const float* pos     = (const float*)d_in[4];
    const float* Wq      = (const float*)d_in[5];
    const float* bq      = (const float*)d_in[6];
    const float* Wk      = (const float*)d_in[7];
    const float* bk      = (const float*)d_in[8];
    const float* Wv      = (const float*)d_in[9];
    const float* bv      = (const float*)d_in[10];
    const float* Wo      = (const float*)d_in[11];
    const float* bo      = (const float*)d_in[12];
    const float* ln1_g   = (const float*)d_in[13];
    const float* ln1_b   = (const float*)d_in[14];
    const float* ln2_g   = (const float*)d_in[15];
    const float* ln2_b   = (const float*)d_in[16];
    const float* ff_w1   = (const float*)d_in[17];
    const float* ff_b1   = (const float*)d_in[18];
    const float* ff_w2   = (const float*)d_in[19];
    const float* ff_b2   = (const float*)d_in[20];
    const float* norm_g  = (const float*)d_in[21];
    const float* norm_b  = (const float*)d_in[22];
    const int*   randat  = (const int*)d_in[23];
    float* out = (float*)d_out;

    char* ws = (char*)d_ws;
    size_t off = 0;
    auto alloc = [&](size_t bytes) -> void* {
        void* p = (void*)(ws + off);
        off += (bytes + 255) & ~(size_t)255;
        return p;
    };

    const size_t DD   = (size_t)768 * 768;
    const size_t DF   = (size_t)768 * 3072;
    const size_t ROWS = 2176;              // 2 * 1088

    unsigned short* wPatch = (unsigned short*)alloc(DD * 2);
    unsigned short* wQb    = (unsigned short*)alloc(12 * DD * 2);
    unsigned short* wKb    = (unsigned short*)alloc(12 * DD * 2);
    unsigned short* wVb    = (unsigned short*)alloc(12 * DD * 2);
    unsigned short* wOb    = (unsigned short*)alloc(12 * DD * 2);
    unsigned short* wF1    = (unsigned short*)alloc(12 * DF * 2);
    unsigned short* wF2    = (unsigned short*)alloc(12 * DF * 2);
    unsigned short* Pm     = (unsigned short*)alloc((size_t)2048 * 768 * 2);
    float*          Xp     = (float*)alloc((size_t)2048 * 768 * 4);
    float*          X      = (float*)alloc(ROWS * 768 * 4);
    unsigned short* xn     = (unsigned short*)alloc(ROWS * 768 * 2);
    unsigned short* Qm     = (unsigned short*)alloc(ROWS * 768 * 2);
    unsigned short* Km     = (unsigned short*)alloc(ROWS * 768 * 2);
    unsigned short* Vm     = (unsigned short*)alloc(ROWS * 768 * 2);
    float*          ctx    = (float*)alloc(ROWS * 768 * 4);
    unsigned short* xn2    = (unsigned short*)alloc(ROWS * 768 * 2);
    unsigned short* H1     = (unsigned short*)alloc(ROWS * 3072 * 2);

    auto cvt = [&](const float* s, unsigned short* d, size_t n) {
        k_convert<<<dim3((unsigned)((n + 255) / 256)), 256, 0, stream>>>(s, d, (int)n);
    };
    cvt(patch_w, wPatch, DD);
    cvt(Wq, wQb, 12 * DD);
    cvt(Wk, wKb, 12 * DD);
    cvt(Wv, wVb, 12 * DD);
    cvt(Wo, wOb, 12 * DD);
    cvt(ff_w1, wF1, 12 * DF);
    cvt(ff_w2, wF2, 12 * DF);

    k_patches<<<dim3(6144), 256, 0, stream>>>(pixel, Pm);

    // patch embedding GEMM: (2048x768) @ (768x768) + patch_b -> Xp (fp32)
    k_gemm<false, false, false, false><<<dim3(6, 16), 256, 0, stream>>>(
        Pm, wPatch, patch_b, Xp, nullptr, 2048, 768, 768);

    k_assemble<<<dim3(6528), 256, 0, stream>>>(Xp, cls, pos, X);

    const dim3 gD(6, 17);     // N=768,  M=2176 (128x128 tiles)
    const dim3 gF(24, 17);    // N=3072, M=2176

    for (int i = 0; i < 12; i++) {
        k_ln_bf16<<<dim3(2176), 256, 0, stream>>>(X, xn, ln1_g + i * 768, ln1_b + i * 768);

        k_gemm<false, true, false, false><<<gD, 256, 0, stream>>>(
            xn, wQb + (size_t)i * DD, bq + i * 768, nullptr, Qm, 2176, 768, 768);
        k_gemm<false, true, false, false><<<gD, 256, 0, stream>>>(
            xn, wKb + (size_t)i * DD, bk + i * 768, nullptr, Km, 2176, 768, 768);
        k_gemm<false, true, false, false><<<gD, 256, 0, stream>>>(
            xn, wVb + (size_t)i * DD, bv + i * 768, nullptr, Vm, 2176, 768, 768);

        k_attn<<<dim3(17, 12, 2), 128, 0, stream>>>(Qm, Km, Vm, randat, ctx);

        // X += ctx @ Wo + bo
        k_gemm<true, false, false, true><<<gD, 256, 0, stream>>>(
            ctx, wOb + (size_t)i * DD, bo + i * 768, X, nullptr, 2176, 768, 768);

        k_ln_bf16<<<dim3(2176), 256, 0, stream>>>(X, xn2, ln2_g + i * 768, ln2_b + i * 768);

        // H1 = gelu(xn2 @ ff_w1 + b1)
        k_gemm<false, true, true, false><<<gF, 256, 0, stream>>>(
            xn2, wF1 + (size_t)i * DF, ff_b1 + i * 3072, nullptr, H1, 2176, 3072, 768);

        // X += H1 @ ff_w2 + b2
        k_gemm<false, false, false, true><<<gD, 256, 0, stream>>>(
            H1, wF2 + (size_t)i * DF, ff_b2 + i * 768, X, nullptr, 2176, 768, 3072);
    }

    k_ln_out<<<dim3(2050), 256, 0, stream>>>(X, out, norm_g, norm_b);
}
